// MultiheadAttention_37701222924793
// MI455X (gfx1250) — compile-verified, matrix operand order checked by reading, not run on hardware
//
#include <hip/hip_runtime.h>

// ---- problem constants (match reference) ----
#define TT   2048
#define BB   2
#define EE   1024
#define HH   16
#define DH   64
#define MM   (TT * BB)          // 4096 rows for the projection GEMMs

typedef __bf16 bf16;
typedef __bf16 v16bf __attribute__((ext_vector_type(16)));
typedef __bf16 v4bf  __attribute__((ext_vector_type(4)));
typedef float  v8f   __attribute__((ext_vector_type(8)));

// D = A*B + C, bf16 inputs, f32 accumulate (V_WMMA_F32_16X16X32_BF16)
__device__ __forceinline__ v8f wmma_bf16(v16bf a, v16bf b, v8f c) {
    return __builtin_amdgcn_wmma_f32_16x16x32_bf16(
        /*neg_a=*/false, a, /*neg_b=*/false, b,
        /*c_mod=*/(short)0, c, /*reuse_a=*/false, /*reuse_b=*/false);
}

// ISA 7.12.2: 16-bit A-matrix 16x32 — element j of lane l holds
// K = 16*(j>>3) + 8*(l>>4) + (j&7), M = l&15.  Same K-map for B (N = l&15).
__device__ __forceinline__ int kbase_of_lane(int lane) { return (lane >> 4) << 3; }

// Fragment from a row-major bf16 matrix: elem j <- p[row*ld + k0 + kmap(j)]
// The two 8-element groups are 16B-contiguous -> two global_load_b128 per lane.
__device__ __forceinline__ v16bf frag_bf16(const bf16* __restrict__ p, int ld,
                                           int row, int k0, int lane) {
    const bf16* a = p + (size_t)row * ld + k0 + kbase_of_lane(lane);
    v16bf f;
#pragma unroll
    for (int j = 0; j < 8; ++j) f[j]     = a[j];
#pragma unroll
    for (int j = 0; j < 8; ++j) f[j + 8] = a[16 + j];
    return f;
}

// butterfly reductions inside each 16-lane half of the wave32
__device__ __forceinline__ float hmax16(float v) {
#pragma unroll
    for (int m = 1; m < 16; m <<= 1) v = fmaxf(v, __shfl_xor(v, m, 32));
    return v;
}
__device__ __forceinline__ float hsum16(float v) {
#pragma unroll
    for (int m = 1; m < 16; m <<= 1) v += __shfl_xor(v, m, 32);
    return v;
}

// ---------------------------------------------------------------------------
// Kernel 0: fp32 -> bf16 conversion (x and the four weight matrices), done
// once so the GEMM hot loops move half the bytes and do zero cvt VALU.
// ---------------------------------------------------------------------------
__global__ void __launch_bounds__(256)
cvt_f32_bf16_kernel(const float* __restrict__ in, bf16* __restrict__ out, int n4) {
    const int i = blockIdx.x * 256 + threadIdx.x;
    if (i < n4) {
        const float4 f = ((const float4*)in)[i];
        v4bf o = { (bf16)f.x, (bf16)f.y, (bf16)f.z, (bf16)f.w };
        ((v4bf*)out)[i] = o;
    }
}

// ---------------------------------------------------------------------------
// Kernel 1: fused QKV projection.  out = (x @ W^T + b) [* 0.125 for q]
// One wave computes a 64x32 tile (4x2 register blocking -> 8 WMMAs per
// k-step vs 6 fragment loads).  Q,K stored [B,H,T,D]; V stored [B,H,D,T]
// so the attention PV-GEMM B-fragments are contiguous.
// ---------------------------------------------------------------------------
__global__ void __launch_bounds__(32)
qkv_proj_kernel(const bf16* __restrict__ x,
                const bf16* __restrict__ Wq, const float* __restrict__ bq,
                const bf16* __restrict__ Wk, const float* __restrict__ bk,
                const bf16* __restrict__ Wv, const float* __restrict__ bv,
                bf16* __restrict__ qo, bf16* __restrict__ ko, bf16* __restrict__ vo) {
    const int lane = threadIdx.x;
    const int m0 = blockIdx.x * 64;
    const int n0 = blockIdx.y * 32;
    const int which = blockIdx.z;

    const bf16*  W    = (which == 0) ? Wq : (which == 1) ? Wk : Wv;
    const float* bias = (which == 0) ? bq : (which == 1) ? bk : bv;
    bf16*        out  = (which == 0) ? qo : (which == 1) ? ko : vo;
    const float scale = (which == 0) ? 0.125f : 1.0f;   // D^-0.5 = 1/8
    const bool  vtrans = (which == 2);

    v8f acc[4][2] = {};
#pragma unroll 2
    for (int kk = 0; kk < EE; kk += 32) {
        v16bf a0 = frag_bf16(x, EE, m0 +  0 + (lane & 15), kk, lane);
        v16bf a1 = frag_bf16(x, EE, m0 + 16 + (lane & 15), kk, lane);
        v16bf a2 = frag_bf16(x, EE, m0 + 32 + (lane & 15), kk, lane);
        v16bf a3 = frag_bf16(x, EE, m0 + 48 + (lane & 15), kk, lane);
        v16bf b0 = frag_bf16(W, EE, n0 +  0 + (lane & 15), kk, lane);
        v16bf b1 = frag_bf16(W, EE, n0 + 16 + (lane & 15), kk, lane);
        acc[0][0] = wmma_bf16(a0, b0, acc[0][0]);
        acc[0][1] = wmma_bf16(a0, b1, acc[0][1]);
        acc[1][0] = wmma_bf16(a1, b0, acc[1][0]);
        acc[1][1] = wmma_bf16(a1, b1, acc[1][1]);
        acc[2][0] = wmma_bf16(a2, b0, acc[2][0]);
        acc[2][1] = wmma_bf16(a2, b1, acc[2][1]);
        acc[3][0] = wmma_bf16(a3, b0, acc[3][0]);
        acc[3][1] = wmma_bf16(a3, b1, acc[3][1]);
        if (kk + 32 < EE) {   // gfx1250 global_prefetch path for next k-block
            __builtin_prefetch(x + (size_t)(m0 + (lane & 15)) * EE + kk + 32, 0, 1);
            __builtin_prefetch(W + (size_t)(n0 + (lane & 15)) * EE + kk + 32, 0, 1);
        }
    }

#pragma unroll
    for (int j = 0; j < 2; ++j) {
        const int n = n0 + j * 16 + (lane & 15);
        const float bn = bias[n];
        const int h = n >> 6, d = n & 63;
#pragma unroll
        for (int i = 0; i < 4; ++i) {
#pragma unroll
            for (int r = 0; r < 8; ++r) {
                const int m = m0 + i * 16 + r + ((lane >> 4) << 3);  // C/D row map
                const int t = m >> 1, bi = m & 1;                    // m = t*B + b
                const float val = (acc[i][j][r] + bn) * scale;
                const size_t idx = vtrans
                    ? (((size_t)bi * HH + h) * DH + d) * TT + t      // V: [B,H,D,T]
                    : (((size_t)bi * HH + h) * TT + t) * DH + d;     // Q,K: [B,H,T,D]
                out[idx] = (bf16)val;
            }
        }
    }
}

// ---------------------------------------------------------------------------
// Kernel 2: flash attention per (b,h).  One wave handles 32 query rows
// (two 16-row A tiles share every K/V fragment), streaming S in chunks of
// 32 with online softmax.  16 WMMAs per chunk.
// q,k: [B*H, T, D] bf16; v: [B*H, D, T] bf16; ctx out: [T*B, E] bf16.
// ---------------------------------------------------------------------------
__global__ void __launch_bounds__(32)
attn_kernel(const bf16* __restrict__ q, const bf16* __restrict__ k,
            const bf16* __restrict__ v, bf16* __restrict__ ctx) {
    const int lane = threadIdx.x;
    const int t0 = blockIdx.x * 32;
    const int bh = blockIdx.y;
    const int bi = bh / HH, h = bh % HH;

    const bf16* qb = q + (size_t)bh * TT * DH;
    const bf16* kb = k + (size_t)bh * TT * DH;
    const bf16* vb = v + (size_t)bh * DH * TT;

    // two Q tiles (each 16x64 = two K=32 A-fragments), register-resident
    v16bf qa[2][2];
#pragma unroll
    for (int qt = 0; qt < 2; ++qt) {
        qa[qt][0] = frag_bf16(qb, DH, t0 + qt * 16 + (lane & 15), 0,  lane);
        qa[qt][1] = frag_bf16(qb, DH, t0 + qt * 16 + (lane & 15), 32, lane);
    }

    v8f c[2][4] = {};                 // ctx accum: 2 q-tiles x (16x64)
    float rmax[2][8], rsum[2][8];
#pragma unroll
    for (int qt = 0; qt < 2; ++qt)
#pragma unroll
        for (int r = 0; r < 8; ++r) { rmax[qt][r] = -1e30f; rsum[qt][r] = 0.0f; }

    __shared__ bf16 plds[32 * 32];    // probs D-layout -> A-layout staging

    for (int s0 = 0; s0 < TT; s0 += 32) {
        // K chunk: two 16-col score tiles x K=64 -> 4 fragments, shared by
        // both q tiles
        v16bf k00 = frag_bf16(kb, DH, s0 +      (lane & 15), 0,  lane);
        v16bf k01 = frag_bf16(kb, DH, s0 +      (lane & 15), 32, lane);
        v16bf k10 = frag_bf16(kb, DH, s0 + 16 + (lane & 15), 0,  lane);
        v16bf k11 = frag_bf16(kb, DH, s0 + 16 + (lane & 15), 32, lane);

#pragma unroll
        for (int qt = 0; qt < 2; ++qt) {
            v8f sA = {}, sB = {};
            sA = wmma_bf16(qa[qt][0], k00, sA);
            sA = wmma_bf16(qa[qt][1], k01, sA);
            sB = wmma_bf16(qa[qt][0], k10, sB);
            sB = wmma_bf16(qa[qt][1], k11, sB);

            // online softmax (row = r + 8*(lane>>4), col = lane&15)
#pragma unroll
            for (int r = 0; r < 8; ++r) {
                float v0 = sA[r], v1 = sB[r];
                float mloc = hmax16(fmaxf(v0, v1));
                float mnew = fmaxf(rmax[qt][r], mloc);
                float corr = __expf(rmax[qt][r] - mnew);
                float p0 = __expf(v0 - mnew);
                float p1 = __expf(v1 - mnew);
                rsum[qt][r] = rsum[qt][r] * corr + hsum16(p0 + p1);
                rmax[qt][r] = mnew;
                c[qt][0][r] *= corr; c[qt][1][r] *= corr;
                c[qt][2][r] *= corr; c[qt][3][r] *= corr;
                const int row = qt * 16 + r + ((lane >> 4) << 3);
                plds[row * 32 +      (lane & 15)] = (bf16)p0;
                plds[row * 32 + 16 + (lane & 15)] = (bf16)p1;
            }
        }
        __syncthreads();   // single-wave WG: DS ops are wave-ordered anyway

        // V chunk 32 x 64: v stored [D,T] so B-frag (N=d per lane, K=s per
        // elem) reads are contiguous b128 loads
        v16bf v0f = frag_bf16(vb, TT, (lane & 15) +  0, s0, lane);
        v16bf v1f = frag_bf16(vb, TT, (lane & 15) + 16, s0, lane);
        v16bf v2f = frag_bf16(vb, TT, (lane & 15) + 32, s0, lane);
        v16bf v3f = frag_bf16(vb, TT, (lane & 15) + 48, s0, lane);

#pragma unroll
        for (int qt = 0; qt < 2; ++qt) {
            v16bf pa;
            const bf16* pp = plds + (size_t)(qt * 16 + (lane & 15)) * 32
                                  + kbase_of_lane(lane);
#pragma unroll
            for (int j = 0; j < 8; ++j) pa[j]     = pp[j];
#pragma unroll
            for (int j = 0; j < 8; ++j) pa[j + 8] = pp[16 + j];

            c[qt][0] = wmma_bf16(pa, v0f, c[qt][0]);
            c[qt][1] = wmma_bf16(pa, v1f, c[qt][1]);
            c[qt][2] = wmma_bf16(pa, v2f, c[qt][2]);
            c[qt][3] = wmma_bf16(pa, v3f, c[qt][3]);
        }
        __syncthreads();
    }

    // normalize and store ctx as bf16 [T*B, E]
#pragma unroll
    for (int qt = 0; qt < 2; ++qt) {
#pragma unroll
        for (int r = 0; r < 8; ++r) {
            const float inv = 1.0f / rsum[qt][r];
            const int t = t0 + qt * 16 + r + ((lane >> 4) << 3);
            const size_t base = ((size_t)t * BB + bi) * EE + h * DH + (lane & 15);
            ctx[base +  0] = (bf16)(c[qt][0][r] * inv);
            ctx[base + 16] = (bf16)(c[qt][1][r] * inv);
            ctx[base + 32] = (bf16)(c[qt][2][r] * inv);
            ctx[base + 48] = (bf16)(c[qt][3][r] * inv);
        }
    }
}

// ---------------------------------------------------------------------------
// Kernel 3: output projection.  out = ctx @ Wo^T + bo  (fp32 out), 64x32
// tile per wave, same 4x2 register blocking.
// ---------------------------------------------------------------------------
__global__ void __launch_bounds__(32)
out_proj_kernel(const bf16* __restrict__ ctx, const bf16* __restrict__ Wo,
                const float* __restrict__ bo, float* __restrict__ out) {
    const int lane = threadIdx.x;
    const int m0 = blockIdx.x * 64;
    const int n0 = blockIdx.y * 32;

    v8f acc[4][2] = {};
#pragma unroll 2
    for (int kk = 0; kk < EE; kk += 32) {
        v16bf a0 = frag_bf16(ctx, EE, m0 +  0 + (lane & 15), kk, lane);
        v16bf a1 = frag_bf16(ctx, EE, m0 + 16 + (lane & 15), kk, lane);
        v16bf a2 = frag_bf16(ctx, EE, m0 + 32 + (lane & 15), kk, lane);
        v16bf a3 = frag_bf16(ctx, EE, m0 + 48 + (lane & 15), kk, lane);
        v16bf b0 = frag_bf16(Wo, EE, n0 +  0 + (lane & 15), kk, lane);
        v16bf b1 = frag_bf16(Wo, EE, n0 + 16 + (lane & 15), kk, lane);
        acc[0][0] = wmma_bf16(a0, b0, acc[0][0]);
        acc[0][1] = wmma_bf16(a0, b1, acc[0][1]);
        acc[1][0] = wmma_bf16(a1, b0, acc[1][0]);
        acc[1][1] = wmma_bf16(a1, b1, acc[1][1]);
        acc[2][0] = wmma_bf16(a2, b0, acc[2][0]);
        acc[2][1] = wmma_bf16(a2, b1, acc[2][1]);
        acc[3][0] = wmma_bf16(a3, b0, acc[3][0]);
        acc[3][1] = wmma_bf16(a3, b1, acc[3][1]);
    }

#pragma unroll
    for (int j = 0; j < 2; ++j) {
        const int n = n0 + j * 16 + (lane & 15);
        const float bn = bo[n];
#pragma unroll
        for (int i = 0; i < 4; ++i) {
#pragma unroll
            for (int r = 0; r < 8; ++r) {
                const int m = m0 + i * 16 + r + ((lane >> 4) << 3);
                out[(size_t)m * EE + n] = acc[i][j][r] + bn;
            }
        }
    }
}

// ---------------------------------------------------------------------------
extern "C" void kernel_launch(void* const* d_in, const int* in_sizes, int n_in,
                              void* d_out, int out_size, void* d_ws, size_t ws_size,
                              hipStream_t stream) {
    const float* x  = (const float*)d_in[0];
    const float* Wq = (const float*)d_in[1];
    const float* bq = (const float*)d_in[2];
    const float* Wk = (const float*)d_in[3];
    const float* bk = (const float*)d_in[4];
    const float* Wv = (const float*)d_in[5];
    const float* bv = (const float*)d_in[6];
    const float* Wo = (const float*)d_in[7];
    const float* bo = (const float*)d_in[8];
    float* out = (float*)d_out;

    // workspace (bf16): x(8M) | Wq,Wk,Wv,Wo(2M each) | q,k,v(8M each) | ctx(8M)
    char* wsb = (char*)d_ws;
    const size_t segX = (size_t)MM * EE * sizeof(bf16);   // 8 MB
    const size_t segW = (size_t)EE * EE * sizeof(bf16);   // 2 MB
    bf16* xbf  = (bf16*)(wsb);
    bf16* wqbf = (bf16*)(wsb + segX);
    bf16* wkbf = (bf16*)(wsb + segX + 1 * segW);
    bf16* wvbf = (bf16*)(wsb + segX + 2 * segW);
    bf16* wobf = (bf16*)(wsb + segX + 3 * segW);
    bf16* qws  = (bf16*)(wsb + segX + 4 * segW);
    bf16* kws  = (bf16*)(wsb + 2 * segX + 4 * segW);
    bf16* vws  = (bf16*)(wsb + 3 * segX + 4 * segW);
    bf16* cws  = (bf16*)(wsb + 4 * segX + 4 * segW);

    const int nx4 = MM * EE / 4, nw4 = EE * EE / 4;
    cvt_f32_bf16_kernel<<<nx4 / 256, 256, 0, stream>>>(x,  xbf,  nx4);
    cvt_f32_bf16_kernel<<<nw4 / 256, 256, 0, stream>>>(Wq, wqbf, nw4);
    cvt_f32_bf16_kernel<<<nw4 / 256, 256, 0, stream>>>(Wk, wkbf, nw4);
    cvt_f32_bf16_kernel<<<nw4 / 256, 256, 0, stream>>>(Wv, wvbf, nw4);
    cvt_f32_bf16_kernel<<<nw4 / 256, 256, 0, stream>>>(Wo, wobf, nw4);

    qkv_proj_kernel<<<dim3(MM / 64, EE / 32, 3), 32, 0, stream>>>(
        xbf, wqbf, bq, wkbf, bk, wvbf, bv, qws, kws, vws);

    attn_kernel<<<dim3(TT / 32, BB * HH), 32, 0, stream>>>(qws, kws, vws, cws);

    out_proj_kernel<<<dim3(MM / 64, EE / 32), 32, 0, stream>>>(cws, wobf, bo, out);
}
